// CONAN_35287451303939
// MI455X (gfx1250) — matrix operand order repaired and checked by reading, not compiled, not run on hardware
//
#include <hip/hip_runtime.h>
#include <hip/hip_bf16.h>

// ---- problem constants ----
#define E_    384
#define CC    32          // codes per visit
#define BB    32          // batch
#define VV    64          // visits
#define HH    6           // heads
#define DD    64          // head dim
#define NSEQ  (BB*VV)     // 2048 sequences
#define NTOK  (NSEQ*CC)   // 65536 token rows
#define TV    10002
#define TVPAD 10016       // padded to multiple of 32

// CDNA5 async global->LDS staging via inline asm (portable across the two
// toolchains; the clang builtin's signature differs between them).
#define CONAN_ASYNC_ASM 1

typedef __attribute__((ext_vector_type(16))) _Float16 v16h;
typedef __attribute__((ext_vector_type(8)))  _Float16 v8h;
typedef __attribute__((ext_vector_type(8)))  float    v8f;

union V16HU { v16h v; v8h h[2]; };

// =====================================================================
// WMMA f16 GEMM:  out[M,N] = act( A[M,K] @ Bw[N,K]^T + bias[N] )
// A,Bw f16 row-major. One wave computes a 32x32 block = 2x2 WMMA tiles
// (each operand fragment reused twice -> half the L2 traffic, 4 wmma
// per K-chunk). Operand layouts per CDNA5 ISA 7.12.2 (wave32):
//   A 16x32:  lane l: m=l&15, g=l>>4; halves 0..7 = K k0+g*8..+7,
//                                     halves 8..15 = K k0+16+g*8..+7
//   B 32x16:  lane l: n=l&15, g=l>>4; halves 0..7 = K k0+g*8..+7, 8..15 = K k0+16+g*8..+7
//   C/D f32:  vgpr r: m = r + 8*(l>>4), n = l&15
// =====================================================================
enum { GEMM_F16 = 0, GEMM_F16_RELU = 1, GEMM_F32 = 2, GEMM_F32_SIGMOID = 3,
       GEMM_F16_TANH = 4 };

__device__ __forceinline__ v16h load_a16x32(const _Float16* __restrict__ Abase,
                                            int lda, int lane, int k0) {
    const _Float16* p = Abase + (size_t)(lane & 15) * lda + k0 + ((lane >> 4) * 8);
    V16HU u;
    u.h[0] = *(const v8h*)(p);
    u.h[1] = *(const v8h*)(p + 16);
    return u.v;
}

__device__ __forceinline__ v16h load_b32x16(const _Float16* __restrict__ Bbase,
                                            int ldb, int lane, int k0) {
    const _Float16* p = Bbase + (size_t)(lane & 15) * ldb + k0 + ((lane >> 4) * 8);
    V16HU u;
    u.h[0] = *(const v8h*)(p);
    u.h[1] = *(const v8h*)(p + 16);
    return u.v;
}

__device__ __forceinline__ void store_tile(v8f acc, void* __restrict__ out, int ldo,
                                           int m0, int n, int Nstore,
                                           const float* __restrict__ bias, int mode)
{
    if (n >= Nstore) return;
    const float bv = bias[n];
    for (int r = 0; r < 8; ++r) {
        float vv = acc[r] + bv;
        size_t idx = (size_t)(m0 + r) * ldo + n;
        if (mode == GEMM_F16) {
            ((_Float16*)out)[idx] = (_Float16)vv;
        } else if (mode == GEMM_F16_RELU) {
            ((_Float16*)out)[idx] = (_Float16)(vv > 0.0f ? vv : 0.0f);
        } else if (mode == GEMM_F32) {
            ((float*)out)[idx] = vv;
        } else if (mode == GEMM_F32_SIGMOID) {
            ((float*)out)[idx] = 1.0f / (1.0f + __expf(-vv));
        } else {
            ((_Float16*)out)[idx] = (_Float16)tanhf(vv);
        }
    }
}

__global__ void conan_gemm_wmma(const _Float16* __restrict__ A, int lda,
                                const _Float16* __restrict__ Bw, int ldb,
                                const float* __restrict__ bias,
                                void* __restrict__ out, int ldo,
                                int M, int Nstore, int tilesN2, int K, int mode)
{
    const int lane   = threadIdx.x & 31;
    const int wave   = threadIdx.x >> 5;
    const int tilesM2 = M >> 5;                       // 32-row blocks
    long wid = (long)blockIdx.x * (blockDim.x >> 5) + wave;
    int tm = (int)(wid % tilesM2);
    int tn = (int)(wid / tilesM2);
    if (tn >= tilesN2) return;

    const _Float16* A0 = A  + (size_t)tm * 32 * lda;
    const _Float16* A1 = A0 + (size_t)16 * lda;
    const _Float16* B0 = Bw + (size_t)tn * 32 * ldb;
    const _Float16* B1 = B0 + (size_t)16 * ldb;

    v8f c00 = {}, c01 = {}, c10 = {}, c11 = {};
    for (int k0 = 0; k0 < K; k0 += 32) {
        v16h a0 = load_a16x32(A0, lda, lane, k0);
        v16h a1 = load_a16x32(A1, lda, lane, k0);
        v16h b0 = load_b32x16(B0, ldb, lane, k0);
        v16h b1 = load_b32x16(B1, ldb, lane, k0);
        c00 = __builtin_amdgcn_wmma_f32_16x16x32_f16(false, a0, false, b0, (short)0, c00, false, false);
        c01 = __builtin_amdgcn_wmma_f32_16x16x32_f16(false, a0, false, b1, (short)0, c01, false, false);
        c10 = __builtin_amdgcn_wmma_f32_16x16x32_f16(false, a1, false, b0, (short)0, c10, false, false);
        c11 = __builtin_amdgcn_wmma_f32_16x16x32_f16(false, a1, false, b1, (short)0, c11, false, false);
    }

    const int nlo = tn * 32 + (lane & 15);
    const int m0  = tm * 32 + ((lane >> 4) * 8);
    store_tile(c00, out, ldo, m0,      nlo,      Nstore, bias, mode);
    store_tile(c01, out, ldo, m0,      nlo + 16, Nstore, bias, mode);
    store_tile(c10, out, ldo, m0 + 16, nlo,      Nstore, bias, mode);
    store_tile(c11, out, ldo, m0 + 16, nlo + 16, Nstore, bias, mode);
}

// ------------------- f32 -> f16 weight conversion (with row pad) -------
__global__ void conan_cvt_pad(const float* __restrict__ src, _Float16* __restrict__ dst,
                              int rows_src, int rows_dst, int cols)
{
    long i = (long)blockIdx.x * blockDim.x + threadIdx.x;
    long total = (long)rows_dst * cols;
    if (i >= total) return;
    int r = (int)(i / cols);
    dst[i] = (r < rows_src) ? (_Float16)src[i] : (_Float16)0.0f;
}

// ------------------- embedding gather to f16 --------------------------
__global__ void conan_embed(const int* __restrict__ ids, const float* __restrict__ emb,
                            _Float16* __restrict__ xh)
{
    long i = (long)blockIdx.x * blockDim.x + threadIdx.x;
    long total = (long)NTOK * E_;
    if (i >= total) return;
    int tok = (int)(i / E_), e = (int)(i % E_);
    xh[i] = (_Float16)emb[(size_t)ids[tok] * E_ + e];
}

// ------------------- per-sequence MHA (f32 VALU) ----------------------
__global__ void conan_attn(const _Float16* __restrict__ qkv, const int* __restrict__ vlen,
                           _Float16* __restrict__ outh)
{
    __shared__ float qs[CC][DD], ks[CC][DD], vs[CC][DD], sc[CC][CC];
    const int s = blockIdx.x;
    const int t = threadIdx.x;      // 256 threads
    const int vl = vlen[s];
    const _Float16* base = qkv + (size_t)s * CC * (3 * E_);

    for (int hd = 0; hd < HH; ++hd) {
        for (int p = t; p < CC * DD; p += 256) {
            int c = p >> 6, d = p & 63;
            const _Float16* row = base + (size_t)c * (3 * E_) + hd * DD + d;
            qs[c][d] = (float)row[0];
            ks[c][d] = (float)row[E_];
            vs[c][d] = (float)row[2 * E_];
        }
        __syncthreads();
        for (int p = t; p < CC * CC; p += 256) {
            int qi = p >> 5, kj = p & 31;
            float acc = 0.0f;
            for (int d = 0; d < DD; ++d) acc += qs[qi][d] * ks[kj][d];
            acc *= 0.125f;                       // 1/sqrt(64)
            if (kj >= vl) acc += -1e9f;
            sc[qi][kj] = acc;
        }
        __syncthreads();
        if (t < CC) {
            float mx = -3.0e38f;
            for (int j = 0; j < CC; ++j) mx = fmaxf(mx, sc[t][j]);
            float sum = 0.0f;
            for (int j = 0; j < CC; ++j) { float e = __expf(sc[t][j] - mx); sc[t][j] = e; sum += e; }
            float inv = 1.0f / sum;
            for (int j = 0; j < CC; ++j) sc[t][j] *= inv;
        }
        __syncthreads();
        for (int p = t; p < CC * DD; p += 256) {
            int qi = p >> 6, d = p & 63;
            float acc = 0.0f;
            for (int j = 0; j < CC; ++j) acc += sc[qi][j] * vs[j][d];
            outh[((size_t)s * CC + qi) * E_ + hd * DD + d] = (_Float16)acc;
        }
        __syncthreads();
    }
}

// ------------------- residual + LayerNorm (wave per row) --------------
__global__ void conan_res_ln(_Float16* __restrict__ xh, const float* __restrict__ proj,
                             const float* __restrict__ g, const float* __restrict__ b,
                             int rows)
{
    const int lane = threadIdx.x & 31;
    const int wave = threadIdx.x >> 5;
    long row = (long)blockIdx.x * (blockDim.x >> 5) + wave;
    if (row >= rows) return;
    _Float16* xr = xh + (size_t)row * E_;
    const float* pr = proj + (size_t)row * E_;
    float vals[E_ / 32];
    float sum = 0.0f, sq = 0.0f;
    for (int j = 0; j < E_ / 32; ++j) {
        int e = lane + j * 32;
        float v = (float)xr[e] + pr[e];
        vals[j] = v; sum += v; sq += v * v;
    }
    for (int o = 16; o > 0; o >>= 1) {
        sum += __shfl_xor(sum, o, 32);
        sq  += __shfl_xor(sq,  o, 32);
    }
    float mean = sum * (1.0f / E_);
    float var  = sq * (1.0f / E_) - mean * mean;
    float inv  = rsqrtf(var + 1e-8f);
    for (int j = 0; j < E_ / 32; ++j) {
        int e = lane + j * 32;
        xr[e] = (_Float16)((vals[j] - mean) * inv * g[e] + b[e]);
    }
}

// ------------------- CLS pooling (f32 + f16 copies) -------------------
__global__ void conan_cls(const _Float16* __restrict__ xh, float* __restrict__ ve,
                          _Float16* __restrict__ veh)
{
    long i = (long)blockIdx.x * blockDim.x + threadIdx.x;
    if (i >= (long)NSEQ * E_) return;
    long r = i / E_, e = i % E_;
    _Float16 v = xh[(size_t)r * CC * E_ + e];   // token 0 of each sequence
    ve[i] = (float)v;
    veh[i] = v;
}

// ------------------- LSTM recurrence (input proj precomputed) ---------
// gx[row, 0:4E] = ve @ Wih^T + bih (from WMMA GEMM). Here: add bhh and
// h@Whh^T, then gate nonlinearities. Block per batch element.
// Gate rows staged into LDS with CDNA5 async global->LDS b128 copies.
__global__ void conan_lstm(const float* __restrict__ gx, const float* __restrict__ Whh,
                           const float* __restrict__ bhh, const int* __restrict__ lengths,
                           float* __restrict__ hsm, _Float16* __restrict__ hsmh)
{
    __shared__ float h[E_], c[E_];
    __shared__ __align__(16) float g[4 * E_];
    const int b = blockIdx.x, t = threadIdx.x;   // 256 threads
    for (int e = t; e < E_; e += 256) { h[e] = 0.0f; c[e] = 0.0f; }
    const int len = lengths[b];
    __syncthreads();
    for (int step = 0; step < VV; ++step) {
        const float* grow = gx + (size_t)(b * VV + step) * (4 * E_);
        // stage the 6KB gate row into LDS
#if CONAN_ASYNC_ASM
        for (int q = t; q < 4 * E_ / 4; q += 256) {   // 384 x b128
            const float* gp = grow + q * 4;           // 16B-aligned global
            unsigned lp = (unsigned)(size_t)(&g[q * 4]); // LDS byte offset (low 32 bits)
            asm volatile("global_load_async_to_lds_b128 %0, %1, off"
                         :: "v"(lp), "v"(gp)
                         : "memory");
        }
        asm volatile("s_wait_asynccnt 0" ::: "memory");
#else
        for (int j = t; j < 4 * E_; j += 256) g[j] = grow[j];
#endif
        __syncthreads();
        for (int j = t; j < 4 * E_; j += 256) {
            float acc = g[j] + bhh[j];
            const float* wh = Whh + (size_t)j * E_;
            for (int k = 0; k < E_; ++k) acc += h[k] * wh[k];
            g[j] = acc;
        }
        __syncthreads();
        const float mask = (step < len) ? 1.0f : 0.0f;
        for (int e = t; e < E_; e += 256) {
            float ig = 1.0f / (1.0f + __expf(-g[e]));
            float fg = 1.0f / (1.0f + __expf(-g[E_ + e]));
            float gt = tanhf(g[2 * E_ + e]);
            float og = 1.0f / (1.0f + __expf(-g[3 * E_ + e]));
            float cn = fg * c[e] + ig * gt;
            float hn = og * tanhf(cn);
            c[e] = cn; h[e] = hn;
            size_t idx = ((size_t)b * VV + step) * E_ + e;
            float hv = hn * mask;
            hsm[idx]  = hv;
            hsmh[idx] = (_Float16)hv;
        }
        __syncthreads();
    }
}

// ------------- u = z @ att2W^T + mask, z = tanh(...) from GEMM --------
__global__ void conan_ured(const _Float16* __restrict__ zh, const float* __restrict__ att2W,
                           const int* __restrict__ lengths, float* __restrict__ u)
{
    const int lane = threadIdx.x & 31;
    const int wave = threadIdx.x >> 5;
    int r = blockIdx.x * (blockDim.x >> 5) + wave;
    if (r >= NSEQ) return;
    const _Float16* z = zh + (size_t)r * E_;
    float s = 0.0f;
    for (int j = 0; j < E_ / 32; ++j) {
        int e = lane + j * 32;
        s += (float)z[e] * att2W[e];
    }
    for (int o = 16; o > 0; o >>= 1) s += __shfl_xor(s, o, 32);
    if (lane == 0) {
        int b = r >> 6, step = r & 63;
        if (step >= lengths[b]) s += -99999.0f;
        u[r] = s;
    }
}

// ------------------- cumulative softmax pooling -> prog (f16) ---------
__global__ void conan_cumul(const float* __restrict__ hsm, const float* __restrict__ u,
                            const int* __restrict__ lengths, _Float16* __restrict__ progh)
{
    __shared__ float ee[VV], den[VV];
    __shared__ int tl;
    const int b = blockIdx.x, t = threadIdx.x;   // 256 threads
    if (t == 0) {
        float mx = -3.0e38f;
        for (int j = 0; j < VV; ++j) mx = fmaxf(mx, u[b * VV + j]);
        float run = 0.0f;
        for (int j = 0; j < VV; ++j) {
            float e = __expf(u[b * VV + j] - mx);
            ee[j] = e; run += e; den[j] = run;
        }
        int m = 0;
        for (int j = 0; j < BB; ++j) m = (lengths[j] > m) ? lengths[j] : m;
        tl = m;   // jnp.max(lengths)
    }
    __syncthreads();
    for (int e = t; e < E_; e += 256) {
        float num = 0.0f;
        for (int j = 0; j < VV; ++j) {
            num += ee[j] * hsm[((size_t)b * VV + j) * E_ + e];
            float val = num / den[j];
            if (j >= tl - 1) val = 0.0f;
            progh[((size_t)b * VV + j) * E_ + e] = (_Float16)val;
        }
    }
}

// =====================================================================
// host side
// =====================================================================
static inline void launch_gemm(hipStream_t s, const _Float16* A, int lda,
                               const _Float16* Bw, int ldb, const float* bias,
                               void* out, int ldo, int M, int Nstore, int Npad,
                               int K, int mode)
{
    int tilesM2 = M / 32, tilesN2 = Npad / 32;
    long waves = (long)tilesM2 * tilesN2;
    int blocks = (int)((waves + 7) / 8);
    conan_gemm_wmma<<<blocks, 256, 0, s>>>(A, lda, Bw, ldb, bias, out, ldo,
                                           M, Nstore, tilesN2, K, mode);
}

static inline void launch_cvt(hipStream_t s, const float* src, _Float16* dst,
                              int rows_src, int rows_dst, int cols)
{
    long total = (long)rows_dst * cols;
    int blocks = (int)((total + 255) / 256);
    conan_cvt_pad<<<blocks, 256, 0, s>>>(src, dst, rows_src, rows_dst, cols);
}

extern "C" void kernel_launch(void* const* d_in, const int* in_sizes, int n_in,
                              void* d_out, int out_size, void* d_ws, size_t ws_size,
                              hipStream_t stream)
{
    (void)in_sizes; (void)n_in; (void)out_size; (void)ws_size;

    const int*   visits  = (const int*)d_in[0];
    const int*   lengths = (const int*)d_in[1];
    const int*   vlens   = (const int*)d_in[2];   // [B*V]
    const float* emb     = (const float*)d_in[3];
    // per-layer param order: Wqkv,bqkv,Wo,bo,ln1g,ln1b,W1,b1,W2,b2,ln2g,ln2b
    const float* L[2][12];
    for (int li = 0; li < 2; ++li)
        for (int j = 0; j < 12; ++j)
            L[li][j] = (const float*)d_in[4 + li * 12 + j];
    const float* lstm_Wih = (const float*)d_in[28];
    const float* lstm_Whh = (const float*)d_in[29];
    const float* lstm_bih = (const float*)d_in[30];
    const float* lstm_bhh = (const float*)d_in[31];
    const float* attW  = (const float*)d_in[32];
    const float* attb  = (const float*)d_in[33];
    const float* att2W = (const float*)d_in[34];
    const float* fcW   = (const float*)d_in[35];
    const float* fcb   = (const float*)d_in[36];
    const float* outW  = (const float*)d_in[37];
    const float* outb  = (const float*)d_in[38];

    // ---- workspace carve-out ----
    char* ws = (char*)d_ws;
    size_t off = 0;
    auto carve = [&](size_t bytes) -> char* {
        off = (off + 255) & ~(size_t)255;
        char* p = ws + off;
        off += bytes;
        return p;
    };
    _Float16* wqkvh[2]; _Float16* woh[2]; _Float16* w1h[2]; _Float16* w2h[2];
    for (int li = 0; li < 2; ++li) {
        wqkvh[li] = (_Float16*)carve((size_t)3 * E_ * E_ * 2);
        woh[li]   = (_Float16*)carve((size_t)E_ * E_ * 2);
        w1h[li]   = (_Float16*)carve((size_t)E_ * E_ * 2);
        w2h[li]   = (_Float16*)carve((size_t)E_ * E_ * 2);
    }
    _Float16* fcwh  = (_Float16*)carve((size_t)E_ * E_ * 2);
    _Float16* attwh = (_Float16*)carve((size_t)E_ * E_ * 2);
    _Float16* wihh  = (_Float16*)carve((size_t)4 * E_ * E_ * 2);
    _Float16* outwh = (_Float16*)carve((size_t)TVPAD * E_ * 2);
    _Float16* xh    = (_Float16*)carve((size_t)NTOK * E_ * 2);
    char*     qkvbuf =            carve((size_t)NTOK * 3 * E_ * 2); // aliased as f32 proj
    _Float16* tmph  = (_Float16*)carve((size_t)NTOK * E_ * 2);
    float*    ve    = (float*)   carve((size_t)NSEQ * E_ * 4);
    _Float16* veh   = (_Float16*)carve((size_t)NSEQ * E_ * 2);
    float*    gx    = (float*)   carve((size_t)NSEQ * 4 * E_ * 4);
    float*    hsm   = (float*)   carve((size_t)NSEQ * E_ * 4);
    _Float16* hsmh  = (_Float16*)carve((size_t)NSEQ * E_ * 2);
    _Float16* zh    = (_Float16*)carve((size_t)NSEQ * E_ * 2);
    float*    uu    = (float*)   carve((size_t)NSEQ * 4);
    _Float16* progh = (_Float16*)carve((size_t)NSEQ * E_ * 2);
    _Float16* fch   = (_Float16*)carve((size_t)NSEQ * E_ * 2);

    _Float16* qkvh  = (_Float16*)qkvbuf;
    float*    projf = (float*)qkvbuf;       // reuse after qkv is consumed

    // ---- convert weights to f16 ----
    for (int li = 0; li < 2; ++li) {
        launch_cvt(stream, L[li][0], wqkvh[li], 3 * E_, 3 * E_, E_);
        launch_cvt(stream, L[li][2], woh[li],   E_, E_, E_);
        launch_cvt(stream, L[li][6], w1h[li],   E_, E_, E_);
        launch_cvt(stream, L[li][8], w2h[li],   E_, E_, E_);
    }
    launch_cvt(stream, fcW,      fcwh,  E_, E_, E_);
    launch_cvt(stream, attW,     attwh, E_, E_, E_);
    launch_cvt(stream, lstm_Wih, wihh,  4 * E_, 4 * E_, E_);
    launch_cvt(stream, outW,     outwh, TV, TVPAD, E_);

    // ---- embedding gather ----
    {
        long total = (long)NTOK * E_;
        conan_embed<<<(int)((total + 255) / 256), 256, 0, stream>>>(visits, emb, xh);
    }

    // ---- 2 encoder layers ----
    for (int li = 0; li < 2; ++li) {
        // qkv = x @ Wqkv^T + bqkv      [NTOK, 3E] f16
        launch_gemm(stream, xh, E_, wqkvh[li], E_, L[li][1], qkvh, 3 * E_,
                    NTOK, 3 * E_, 3 * E_, E_, GEMM_F16);
        // multi-head attention -> tmph
        conan_attn<<<NSEQ, 256, 0, stream>>>(qkvh, vlens, tmph);
        // proj = attn @ Wo^T + bo      [NTOK, E] f32   (overwrites qkv region)
        launch_gemm(stream, tmph, E_, woh[li], E_, L[li][3], projf, E_,
                    NTOK, E_, E_, E_, GEMM_F32);
        // x = LN(x + proj)
        conan_res_ln<<<NTOK / 8, 256, 0, stream>>>(xh, projf, L[li][4], L[li][5], NTOK);
        // ff1 = relu(x @ W1^T + b1)    f16
        launch_gemm(stream, xh, E_, w1h[li], E_, L[li][7], tmph, E_,
                    NTOK, E_, E_, E_, GEMM_F16_RELU);
        // ff2 = ff1 @ W2^T + b2        f32
        launch_gemm(stream, tmph, E_, w2h[li], E_, L[li][9], projf, E_,
                    NTOK, E_, E_, E_, GEMM_F32);
        // x = LN(x + ff2)
        conan_res_ln<<<NTOK / 8, 256, 0, stream>>>(xh, projf, L[li][10], L[li][11], NTOK);
    }

    // ---- CLS pooling ----
    {
        long total = (long)NSEQ * E_;
        conan_cls<<<(int)((total + 255) / 256), 256, 0, stream>>>(xh, ve, veh);
    }

    // ---- LSTM: input projection on matrix cores, recurrence on VALU ----
    launch_gemm(stream, veh, E_, wihh, E_, lstm_bih, gx, 4 * E_,
                NSEQ, 4 * E_, 4 * E_, E_, GEMM_F32);
    conan_lstm<<<BB, 256, 0, stream>>>(gx, lstm_Whh, lstm_bhh, lengths, hsm, hsmh);

    // ---- attention-pool: z = tanh(out @ attW^T + attb) on matrix cores ----
    launch_gemm(stream, hsmh, E_, attwh, E_, attb, zh, E_,
                NSEQ, E_, E_, E_, GEMM_F16_TANH);
    conan_ured<<<NSEQ / 8, 256, 0, stream>>>(zh, att2W, lengths, uu);
    conan_cumul<<<BB, 256, 0, stream>>>(hsm, uu, lengths, progh);

    // ---- head: relu(prog @ fcW^T + fcb) @ outW^T + outb -> sigmoid ----
    launch_gemm(stream, progh, E_, fcwh, E_, fcb, fch, E_,
                NSEQ, E_, E_, E_, GEMM_F16_RELU);
    launch_gemm(stream, fch, E_, outwh, E_, outb, (float*)d_out, TV,
                NSEQ, TV, TVPAD, E_, GEMM_F32_SIGMOID);
}
